// MoEMLP_59639915872321
// MI455X (gfx1250) — matrix operand order repaired and checked by reading, not compiled, not run on hardware
//
#include <hip/hip_runtime.h>
#include <hip/hip_bf16.h>
#include <hip/hip_fp16.h>

// MoE MLP: LayerNorm -> router(top-4 softmax gate) -> 8-expert FFN (GELU) -> gated sum.
// f16 WMMA (v_wmma_f32_16x16x32_f16), f32 accumulate. Weights pre-transposed to f16
// column-major in workspace (L2-resident). 64-token tiles + 16-wave workgroups give
// 2-4x register reuse of B fragments (operand-bandwidth roofline, not FLOP roofline,
// is the binding constraint at 16-token tiles).

#define T_TOK   32768
#define D_IN    1024
#define H_DIM   2048
#define OUT_DIM 1024
#define NE      8
#define TOPK    4
#define EPSV    1e-5f

typedef __attribute__((ext_vector_type(16))) _Float16 v16h;
typedef __attribute__((ext_vector_type(8)))  _Float16 v8h;
typedef __attribute__((ext_vector_type(8)))  float    v8f;

__device__ __forceinline__ v16h frag_cat(v8h lo, v8h hi) {
  return __builtin_shufflevector(lo, hi, 0,1,2,3,4,5,6,7,8,9,10,11,12,13,14,15);
}

// A-fragment: 16x32 f16 from row-major buffer (leading dim `ld` halfs).  (ISA 7.12.2)
__device__ __forceinline__ v16h load_a(const _Float16* base, int ld, int kBase, int lane) {
  const _Float16* p = base + (lane & 15) * ld + kBase + ((lane >> 4) << 3);
  v8h lo = *(const v8h*)(p);
  v8h hi = *(const v8h*)(p + 16);
  return frag_cat(lo, hi);
}

// B-fragment: 32x16 f16 from column-major weights wcols[n][k], leading dim `ld`.
__device__ __forceinline__ v16h load_b(const _Float16* wcols, int ld, int nBase, int kBase, int lane) {
  const _Float16* p = wcols + (size_t)(nBase + (lane & 15)) * ld + kBase + ((lane >> 4) << 3);
  v8h lo = *(const v8h*)(p);
  v8h hi = *(const v8h*)(p + 16);
  return frag_cat(lo, hi);
}

// ---------------------------------------------------------------------------
// Kernel 0: f32 [R][C] -> f16 [C][R] transpose+convert (per expert in grid.z).
// ---------------------------------------------------------------------------
__global__ __launch_bounds__(256) void transpose_cvt_kernel(
    const float* __restrict__ in, _Float16* __restrict__ out, int R, int C)
{
  __shared__ float tile[32][33];
  const size_t mat = (size_t)R * C;
  const float* ip = in + (size_t)blockIdx.z * mat;
  _Float16*    op = out + (size_t)blockIdx.z * mat;
  const int c0 = blockIdx.x * 32, r0 = blockIdx.y * 32;
  const int tx = threadIdx.x, ty = threadIdx.y;   // block (32, 8)
  #pragma unroll
  for (int i = 0; i < 4; ++i)
    tile[ty + i * 8][tx] = ip[(size_t)(r0 + ty + i * 8) * C + (c0 + tx)];
  __syncthreads();
  #pragma unroll
  for (int i = 0; i < 4; ++i)
    op[(size_t)(c0 + ty + i * 8) * R + (r0 + tx)] = (_Float16)tile[tx][ty + i * 8];
}

// ---------------------------------------------------------------------------
// Kernel 1: LayerNorm + router + top-4 softmax gates. One wave32 per token.
// ---------------------------------------------------------------------------
__global__ __launch_bounds__(256) void ln_router_kernel(
    const float* __restrict__ x, const float* __restrict__ gamma, const float* __restrict__ beta,
    const float* __restrict__ rw, const float* __restrict__ rb,
    _Float16* __restrict__ xn16, float* __restrict__ gates)
{
  const int lane = threadIdx.x & 31;
  const int t = blockIdx.x * 8 + (threadIdx.x >> 5);
  const float* xr = x + (size_t)t * D_IN;

  float4 v[8];
  float s = 0.f, s2 = 0.f;
  #pragma unroll
  for (int i = 0; i < 8; ++i) {
    v[i] = *(const float4*)(xr + lane * 32 + i * 4);
    s  += v[i].x + v[i].y + v[i].z + v[i].w;
    s2 += v[i].x * v[i].x + v[i].y * v[i].y + v[i].z * v[i].z + v[i].w * v[i].w;
  }
  #pragma unroll
  for (int m = 16; m >= 1; m >>= 1) {
    s  += __shfl_xor(s,  m, 32);
    s2 += __shfl_xor(s2, m, 32);
  }
  const float mu   = s * (1.0f / D_IN);
  const float var  = s2 * (1.0f / D_IN) - mu * mu;
  const float rstd = rsqrtf(var + EPSV);

  float p[NE] = {0.f, 0.f, 0.f, 0.f, 0.f, 0.f, 0.f, 0.f};
  _Float16 xo[32];
  #pragma unroll
  for (int i = 0; i < 8; ++i) {
    const float xv4[4] = {v[i].x, v[i].y, v[i].z, v[i].w};
    #pragma unroll
    for (int j = 0; j < 4; ++j) {
      const int d = lane * 32 + i * 4 + j;
      const float xn = (xv4[j] - mu) * rstd * gamma[d] + beta[d];
      xo[i * 4 + j] = (_Float16)xn;
      const float4 r0 = *(const float4*)(rw + d * NE);
      const float4 r1 = *(const float4*)(rw + d * NE + 4);
      p[0] += xn * r0.x; p[1] += xn * r0.y; p[2] += xn * r0.z; p[3] += xn * r0.w;
      p[4] += xn * r1.x; p[5] += xn * r1.y; p[6] += xn * r1.z; p[7] += xn * r1.w;
    }
  }
  _Float16* xd = xn16 + (size_t)t * D_IN + lane * 32;
  #pragma unroll
  for (int i = 0; i < 4; ++i) *(v8h*)(xd + i * 8) = *(const v8h*)(xo + i * 8);

  #pragma unroll
  for (int e = 0; e < NE; ++e) {
    #pragma unroll
    for (int m = 16; m >= 1; m >>= 1) p[e] += __shfl_xor(p[e], m, 32);
  }

  if (lane == 0) {
    float lg[NE];
    #pragma unroll
    for (int e = 0; e < NE; ++e) lg[e] = p[e] + rb[e];
    bool used[NE] = {false};
    float tv[TOPK]; int ti[TOPK];
    for (int k = 0; k < TOPK; ++k) {
      float best = -3.4e38f; int bi = 0;
      for (int e = 0; e < NE; ++e)
        if (!used[e] && lg[e] > best) { best = lg[e]; bi = e; }
      used[bi] = true; tv[k] = best; ti[k] = bi;
    }
    const float mx = tv[0];
    float ex[TOPK], sum = 0.f;
    for (int k = 0; k < TOPK; ++k) { ex[k] = expf(tv[k] - mx); sum += ex[k]; }
    float go[NE] = {0.f, 0.f, 0.f, 0.f, 0.f, 0.f, 0.f, 0.f};
    for (int k = 0; k < TOPK; ++k) go[ti[k]] = ex[k] / sum;
    for (int e = 0; e < NE; ++e) gates[(size_t)t * NE + e] = go[e];
  }
}

// ---------------------------------------------------------------------------
// Kernel 2: fused expert FFN. One workgroup = 16 wave32 (512 thr) per 64 tokens.
// Per expert, per H-chunk of 128:
//   GEMM1: wave w -> N-subtile (w&7), M-subtiles {2*(w>>3), +1}; B reused 2x.
//   GEMM2: wave w -> 64 out cols (4 N-subtiles), all 4 M-subtiles; B reused 4x,
//          A reused 4x; 16 accumulator tiles (128 VGPRs).
// Cross-expert y accumulation via workgroup-private global RMW (no atomics).
// ---------------------------------------------------------------------------
#define TM    64
#define MS    4            // M (token) subtiles per tile
#define HC    128
#define XN_LD (D_IN + 8)   // 1032 halfs: row stride 516 dwords -> conflict-free
#define HS_LD (HC + 8)     // 136 halfs

__global__ __launch_bounds__(512) void moe_ffn_kernel(
    const _Float16* __restrict__ xn16, const float* __restrict__ gates,
    const _Float16* __restrict__ w1t,  const float* __restrict__ b1,
    const _Float16* __restrict__ w2t,  const float* __restrict__ b2,
    float* __restrict__ y)
{
  __shared__ __align__(16) _Float16 xn_s[TM * XN_LD];  // 129 KB
  __shared__ __align__(16) _Float16 h_s[TM * HS_LD];   // 17 KB
  __shared__ float gate_s[TM * NE];                    // 2 KB

  const int tid  = threadIdx.x;
  const int wv   = tid >> 5;        // 0..15
  const int lane = tid & 31;
  const int t0   = blockIdx.x * TM;

  // Stage activation tile (64 x 1024 f16) into LDS as b128 chunks.
  {
    const _Float16* src = xn16 + (size_t)t0 * D_IN;
    const int chunks = TM * (D_IN / 8);   // 8192
    for (int c = tid; c < chunks; c += 512) {
      const int r = c / (D_IN / 8);
      const int col = (c % (D_IN / 8)) * 8;
      *(v8h*)(xn_s + r * XN_LD + col) = *(const v8h*)(src + r * D_IN + col);
    }
    gate_s[tid] = gates[(size_t)t0 * NE + tid];   // TM*NE == 512
  }
  __syncthreads();

  bool wrote = false;
  for (int e = 0; e < NE; ++e) {
    // Skip expert if no token in this tile routed to it (uniform branch).
    bool active = false;
    for (int m = 0; m < TM; ++m) active |= (gate_s[m * NE + e] != 0.0f);
    if (!active) continue;

    const _Float16* w1cols = w1t + (size_t)e * H_DIM * D_IN;    // [H][D]
    const _Float16* w2cols = w2t + (size_t)e * OUT_DIM * H_DIM; // [OUT][H]
    v8f yacc[16] = {};   // [os(4)][ms(4)]

    for (int ch = 0; ch < H_DIM / HC; ++ch) {
      const int hBase = ch * HC;

      // ---- GEMM1: N-subtile (w&7), M-subtiles mg..mg+1, full K = D ----
      const int n1 = hBase + (wv & 7) * 16;
      const int mg = (wv >> 3) * 2;
      v8f c1[2] = {};
      #pragma unroll 4
      for (int k = 0; k < D_IN; k += 32) {
        v16h b = load_b(w1cols, D_IN, n1, k, lane);
        #pragma unroll
        for (int i = 0; i < 2; ++i) {
          v16h a = load_a(xn_s + (mg + i) * 16 * XN_LD, XN_LD, k, lane);
          c1[i] = __builtin_amdgcn_wmma_f32_16x16x32_f16(false, a, false, b,
                                                         (short)0, c1[i], false, false);
        }
      }
      __syncthreads();  // previous chunk's GEMM2 reads of h_s done

      // bias + exact GELU -> h_s (f16)
      {
        const int nc = (wv & 7) * 16 + (lane & 15);
        const float b1v = b1[(size_t)e * H_DIM + hBase + nc];
        #pragma unroll
        for (int i = 0; i < 2; ++i) {
          #pragma unroll
          for (int r = 0; r < 8; ++r) {
            const int m = (mg + i) * 16 + r + ((lane >> 4) << 3);
            const float vz = c1[i][r] + b1v;
            const float g = 0.5f * vz * (1.0f + erff(vz * 0.70710678118654752440f));
            h_s[m * HS_LD + nc] = (_Float16)g;
          }
        }
      }
      __syncthreads();  // h chunk ready

      // ---- GEMM2: out cols [wv*64, +64), all 4 M-subtiles, K = this chunk ----
      #pragma unroll
      for (int kk = 0; kk < HC / 32; ++kk) {
        v16h a2[MS];
        #pragma unroll
        for (int ms = 0; ms < MS; ++ms)
          a2[ms] = load_a(h_s + ms * 16 * HS_LD, HS_LD, kk * 32, lane);
        #pragma unroll
        for (int os = 0; os < 4; ++os) {
          v16h b = load_b(w2cols, H_DIM, wv * 64 + os * 16, hBase + kk * 32, lane);
          #pragma unroll
          for (int ms = 0; ms < MS; ++ms)
            yacc[os * MS + ms] = __builtin_amdgcn_wmma_f32_16x16x32_f16(
                false, a2[ms], false, b, (short)0, yacc[os * MS + ms], false, false);
        }
      }
    }

    // y(RMW) += gate[m] * (acc + b2[n]);  WG exclusively owns its 64 token rows.
    #pragma unroll
    for (int os = 0; os < 4; ++os) {
      const int n = wv * 64 + os * 16 + (lane & 15);
      const float b2v = b2[(size_t)e * OUT_DIM + n];
      #pragma unroll
      for (int ms = 0; ms < MS; ++ms) {
        #pragma unroll
        for (int r = 0; r < 8; ++r) {
          const int m = ms * 16 + r + ((lane >> 4) << 3);
          const float val = gate_s[m * NE + e] * (yacc[os * MS + ms][r] + b2v);
          const size_t idx = (size_t)(t0 + m) * OUT_DIM + n;
          if (wrote) y[idx] += val; else y[idx] = val;
        }
      }
    }
    wrote = true;
  }

  if (!wrote) {
    #pragma unroll
    for (int os = 0; os < 4; ++os) {
      const int n = wv * 64 + os * 16 + (lane & 15);
      #pragma unroll
      for (int ms = 0; ms < MS; ++ms)
        #pragma unroll
        for (int r = 0; r < 8; ++r) {
          const int m = ms * 16 + r + ((lane >> 4) << 3);
          y[(size_t)(t0 + m) * OUT_DIM + n] = 0.0f;
        }
    }
  }
}

// ---------------------------------------------------------------------------
// Launch. Inputs: x, ln_gamma, ln_beta, router_w, router_b, w1, b1, w2, b2.
// Workspace: xn16 (64MiB) | gates (1MiB) | w1t f16 (32MiB) | w2t f16 (32MiB).
// ---------------------------------------------------------------------------
extern "C" void kernel_launch(void* const* d_in, const int* in_sizes, int n_in,
                              void* d_out, int out_size, void* d_ws, size_t ws_size,
                              hipStream_t stream) {
  (void)in_sizes; (void)n_in; (void)out_size; (void)ws_size;
  const float* x    = (const float*)d_in[0];
  const float* ln_g = (const float*)d_in[1];
  const float* ln_b = (const float*)d_in[2];
  const float* rw   = (const float*)d_in[3];
  const float* rb   = (const float*)d_in[4];
  const float* w1   = (const float*)d_in[5];
  const float* b1   = (const float*)d_in[6];
  const float* w2   = (const float*)d_in[7];
  const float* b2   = (const float*)d_in[8];
  float* y = (float*)d_out;

  char* ws = (char*)d_ws;
  _Float16* xn16  = (_Float16*)ws;
  float*    gates = (float*)(ws + (size_t)T_TOK * D_IN * 2);
  _Float16* w1t   = (_Float16*)(ws + (size_t)T_TOK * D_IN * 2 + (size_t)T_TOK * NE * 4);
  _Float16* w2t   = w1t + (size_t)NE * (size_t)D_IN * H_DIM;

  dim3 tb(32, 8, 1);
  transpose_cvt_kernel<<<dim3(H_DIM / 32, D_IN / 32, NE), tb, 0, stream>>>(w1, w1t, D_IN, H_DIM);
  transpose_cvt_kernel<<<dim3(OUT_DIM / 32, H_DIM / 32, NE), tb, 0, stream>>>(w2, w2t, H_DIM, OUT_DIM);
  ln_router_kernel<<<T_TOK / 8, 256, 0, stream>>>(x, ln_g, ln_b, rw, rb, xn16, gates);
  moe_ffn_kernel<<<T_TOK / TM, 512, 0, stream>>>(xn16, gates, w1t, b1, w2t, b2, y);
}
